// EGraphSAGE_8297876816047
// MI455X (gfx1250) — matrix operand order repaired and checked by reading, not compiled, not run on hardware
//
#include <hip/hip_runtime.h>
#include <hip/hip_bf16.h>

// ---------------------------------------------------------------------------
// EGraphSAGE (2-layer) for MI455X / gfx1250: wave32, bf16 WMMA GEMMs,
// async global->LDS staging, fp32 L2 atomics for segment-sum.
// ---------------------------------------------------------------------------

typedef __attribute__((ext_vector_type(16))) __bf16 v16bf;
typedef __attribute__((ext_vector_type(8)))  float  v8f;
typedef __attribute__((ext_vector_type(2)))  __bf16 bf16x2;
typedef int v4i __attribute__((vector_size(16)));

__device__ __forceinline__ void atomAddF(float* p, float v) {
    __hip_atomic_fetch_add(p, v, __ATOMIC_RELAXED, __HIP_MEMORY_SCOPE_AGENT);
}

// ---- async global->LDS (gfx1250 ASYNCcnt path), with sync fallback --------
#if defined(__has_builtin)
#if __has_builtin(__builtin_amdgcn_global_load_async_to_lds_b128)
#define ASYNC_LDS 1
#endif
#endif
#ifndef ASYNC_LDS
#define ASYNC_LDS 0
#endif

__device__ __forceinline__ void cp16_g2lds(void* lds, const void* g) {
#if ASYNC_LDS
    __builtin_amdgcn_global_load_async_to_lds_b128(
        (__attribute__((address_space(1))) v4i*)(uintptr_t)(g),
        (__attribute__((address_space(3))) v4i*)(lds), 0, 0);
#else
    *reinterpret_cast<uint4*>(lds) = *reinterpret_cast<const uint4*>(g);
#endif
}

__device__ __forceinline__ void cp_wait_all() {
#if ASYNC_LDS
#if __has_builtin(__builtin_amdgcn_s_wait_asynccnt)
    __builtin_amdgcn_s_wait_asynccnt(0);
#else
    asm volatile("s_wait_asynccnt 0x0" ::: "memory");
#endif
#endif
}

// ---- WMMA fragment loaders (CDNA5 wave32 layouts, CDNA5 ISA) ----
// A (16x32 bf16): lane0-15 -> M=lane; lanes16-31 -> same M, K+8.
//   VGPR v<4: K base 2v ; v>=4: 16+2(v-4); pair (K,K+1) in one dword.
__device__ __forceinline__ v16bf load_a_frag(const __bf16* sA, int m0, int k0,
                                             int ldk, int lane) {
    v16bf a;
    const int m  = m0 + (lane & 15);
    const int hi = (lane >> 4) & 1;
    const __bf16* row = sA + m * ldk + k0 + (hi ? 8 : 0);
#pragma unroll
    for (int vv = 0; vv < 8; ++vv) {
        const int kb = (vv < 4) ? (2 * vv) : (16 + 2 * (vv - 4));
        bf16x2 p = *reinterpret_cast<const bf16x2*>(row + kb);
        a[2 * vv]     = p.x;
        a[2 * vv + 1] = p.y;
    }
    return a;
}

// B (32x16 bf16): lane0-15 -> N=lane; lanes16-31 -> same N, K+8.
//   VGPR v<4: K base 2v ; v>=4: 16+2(v-4). Weights staged K-contiguous per column.
__device__ __forceinline__ v16bf load_b_frag(const __bf16* sW, int n0, int k0,
                                             int ldk, int lane) {
    v16bf b;
    const int n  = n0 + (lane & 15);
    const int hi = (lane >> 4) & 1;
    const __bf16* col = sW + n * ldk + k0 + (hi ? 8 : 0);
#pragma unroll
    for (int vv = 0; vv < 8; ++vv) {
        const int kb = (vv < 4) ? (2 * vv) : (16 + 2 * (vv - 4));
        bf16x2 p = *reinterpret_cast<const bf16x2*>(col + kb);
        b[2 * vv]     = p.x;
        b[2 * vv + 1] = p.y;
    }
    return b;
}

// C/D (16x16 f32): VGPR r -> M = r + 8*(lane>=16), N = lane%16.
// Optionally mirrors the relu output into a bf16 buffer (for later gathers).
__device__ __forceinline__ void store_c_relu(float* out, __bf16* outb,
                                             int row0, int col0, const v8f& c,
                                             const float* bias, int rowMax,
                                             int lane) {
    const int n  = col0 + (lane & 15);
    const int mo = (lane >> 4) ? 8 : 0;
    const float bv = bias[n];
#pragma unroll
    for (int r = 0; r < 8; ++r) {
        const int m = row0 + mo + r;
        if (m < rowMax) {
            const float val = fmaxf(c[r] + bv, 0.0f);
            out[(size_t)m * 128 + n] = val;
            if (outb) outb[(size_t)m * 128 + n] = (__bf16)val;
        }
    }
}

#define WMMA_BF16(a, b, c)                                                    \
    __builtin_amdgcn_wmma_f32_16x16x32_bf16(false, (a), false, (b), (short)0, \
                                            (c), false, false)

// ---------------------------------------------------------------------------
// Utility kernels
// ---------------------------------------------------------------------------
__global__ void zero_kernel(float* __restrict__ p, long n) {
    for (long i = blockIdx.x * (long)blockDim.x + threadIdx.x; i < n;
         i += (long)gridDim.x * blockDim.x)
        p[i] = 0.0f;
}

__global__ void deg_kernel(const int* __restrict__ v, float* __restrict__ deg,
                           int E) {
    for (int e = blockIdx.x * blockDim.x + threadIdx.x; e < E;
         e += gridDim.x * blockDim.x)
        atomAddF(&deg[v[e]], 1.0f);
}

// W [K][128] fp32 row-major  ->  Wt [128][K] bf16 (column-contiguous for B).
__global__ void convert_wt_kernel(const float* __restrict__ W,
                                  __bf16* __restrict__ Wt, int K) {
    for (int idx = blockIdx.x * blockDim.x + threadIdx.x; idx < K * 128;
         idx += gridDim.x * blockDim.x) {
        const int k = idx >> 7, n = idx & 127;
        Wt[n * K + k] = (__bf16)W[idx];
    }
}

// segment-sum of concat(nsrc[u[e]], esrc[e]) onto msum[v[e]] (width Ws+We).
__global__ void scatter_kernel(const float* __restrict__ nsrc, int Ws,
                               const float* __restrict__ esrc, int We,
                               const int* __restrict__ u,
                               const int* __restrict__ v,
                               float* __restrict__ msum, int E) {
    const int Wm  = Ws + We;
    const int tpe = Wm >> 2;  // threads per edge (float4 each)
    const long total = (long)E * tpe;
    for (long t = blockIdx.x * (long)blockDim.x + threadIdx.x; t < total;
         t += (long)gridDim.x * blockDim.x) {
        const int e = (int)(t / tpe);
        const int c = (int)(t % tpe) * 4;
        const int vv = v[e];
        const float* src = (c < Ws)
                               ? &nsrc[(size_t)u[e] * Ws + c]
                               : &esrc[(size_t)e * We + (c - Ws)];
        const float4 val = *reinterpret_cast<const float4*>(src);
        float* dst = &msum[(size_t)vv * Wm + c];
        atomAddF(dst + 0, val.x);
        atomAddF(dst + 1, val.y);
        atomAddF(dst + 2, val.z);
        atomAddF(dst + 3, val.w);
    }
}

// ---------------------------------------------------------------------------
// Node-apply GEMM: out = relu([hprev | msum/deg] @ W + b), W is K x 128.
// 32-row x 128-col block tile, 8 waves; B fragment shared across row tiles.
// ---------------------------------------------------------------------------
#define NG_ROWS 32
#define NG_MAXK 384

__global__ __launch_bounds__(256) void node_gemm_kernel(
    const float* __restrict__ hprev, int K1, const float* __restrict__ msum,
    int K2, const float* __restrict__ deg, const __bf16* __restrict__ Wt,
    const float* __restrict__ bias, float* __restrict__ out,
    __bf16* __restrict__ outb, int N) {
    __shared__ __attribute__((aligned(16))) __bf16 sA[NG_ROWS * NG_MAXK];
    __shared__ __attribute__((aligned(16))) __bf16 sW[128 * NG_MAXK];
    __shared__ float sRdeg[NG_ROWS];

    const int K    = K1 + K2;
    const int row0 = blockIdx.x * NG_ROWS;
    const int tid  = threadIdx.x;

    // weights: bf16 [128][K] -> LDS, 16B chunks (async path)
    for (int ch = tid; ch < K * 16; ch += 256)
        cp16_g2lds(&sW[ch * 8], &Wt[ch * 8]);

    if (tid < NG_ROWS) {
        const int r = row0 + tid;
        sRdeg[tid] = (r < N) ? (1.0f / fmaxf(deg[r], 1.0f)) : 0.0f;
    }
    __syncthreads();
    // A tile: fused concat + mean-normalization, fp32 -> bf16
    for (int idx = tid; idx < NG_ROWS * K; idx += 256) {
        const int r = idx / K, k = idx - r * K;
        const int row = row0 + r;
        float val = 0.0f;
        if (row < N)
            val = (k < K1) ? hprev[(size_t)row * K1 + k]
                           : msum[(size_t)row * K2 + (k - K1)] * sRdeg[r];
        sA[r * K + k] = (__bf16)val;
    }
    cp_wait_all();
    __syncthreads();

    const int wave = tid >> 5, lane = tid & 31;
    const int n0 = wave * 16;
    v8f c0 = {}, c1 = {};
    for (int k0 = 0; k0 < K; k0 += 32) {
        v16bf b  = load_b_frag(sW, n0, k0, K, lane);
        v16bf a0 = load_a_frag(sA, 0, k0, K, lane);
        v16bf a1 = load_a_frag(sA, 16, k0, K, lane);
        c0 = WMMA_BF16(a0, b, c0);
        c1 = WMMA_BF16(a1, b, c1);
    }
    store_c_relu(out, outb, row0, n0, c0, bias, N, lane);
    store_c_relu(out, outb, row0 + 16, n0, c1, bias, N, lane);
}

// ---------------------------------------------------------------------------
// Edge GEMM: out[e] = relu([hn[u[e]] | hn[v[e]]] @ W + b), K=256, N=128.
// A rows gathered as bf16 straight from hnb (L2-resident) via async copies.
// 64 edges/block; B fragment shared across 4 row tiles.
// ---------------------------------------------------------------------------
#define EG_EDGES 64

__global__ __launch_bounds__(256) void edge_gemm_kernel(
    const __bf16* __restrict__ hnb, const int* __restrict__ u,
    const int* __restrict__ v, const __bf16* __restrict__ Wt,
    const float* __restrict__ bias, float* __restrict__ out, int E) {
    __shared__ __attribute__((aligned(16))) __bf16 sA[EG_EDGES * 256];
    __shared__ __attribute__((aligned(16))) __bf16 sW[128 * 256];

    const int e0  = blockIdx.x * EG_EDGES;
    const int tid = threadIdx.x;

    // weights: 128x256 bf16 = 64KB = 4096 x 16B chunks
    for (int ch = tid; ch < 4096; ch += 256)
        cp16_g2lds(&sW[ch * 8], &Wt[ch * 8]);
    // A: 64 rows x 16 chunks of 8 bf16; first 8 chunks from u-row, rest v-row
    for (int ch = tid; ch < EG_EDGES * 16; ch += 256) {
        const int r = ch >> 4, cc = ch & 15;
        const int e = e0 + r;
        __bf16* dst = &sA[r * 256 + cc * 8];
        if (e < E) {
            const int nd = (cc < 8) ? u[e] : v[e];
            cp16_g2lds(dst, &hnb[(size_t)nd * 128 + (cc & 7) * 8]);
        } else {
            *reinterpret_cast<uint4*>(dst) = make_uint4(0u, 0u, 0u, 0u);
        }
    }
    cp_wait_all();
    __syncthreads();

    const int wave = tid >> 5, lane = tid & 31;
    const int n0 = wave * 16;
    v8f c0 = {}, c1 = {}, c2 = {}, c3 = {};
#pragma unroll
    for (int k0 = 0; k0 < 256; k0 += 32) {
        v16bf b  = load_b_frag(sW, n0, k0, 256, lane);
        v16bf a0 = load_a_frag(sA, 0, k0, 256, lane);
        v16bf a1 = load_a_frag(sA, 16, k0, 256, lane);
        v16bf a2 = load_a_frag(sA, 32, k0, 256, lane);
        v16bf a3 = load_a_frag(sA, 48, k0, 256, lane);
        c0 = WMMA_BF16(a0, b, c0);
        c1 = WMMA_BF16(a1, b, c1);
        c2 = WMMA_BF16(a2, b, c2);
        c3 = WMMA_BF16(a3, b, c3);
    }
    store_c_relu(out, nullptr, e0, n0, c0, bias, E, lane);
    store_c_relu(out, nullptr, e0 + 16, n0, c1, bias, E, lane);
    store_c_relu(out, nullptr, e0 + 32, n0, c2, bias, E, lane);
    store_c_relu(out, nullptr, e0 + 48, n0, c3, bias, E, lane);
}

// ---------------------------------------------------------------------------
// Host-side orchestration (graph-capture safe: launches only).
// ---------------------------------------------------------------------------
extern "C" void kernel_launch(void* const* d_in, const int* in_sizes, int n_in,
                              void* d_out, int out_size, void* d_ws,
                              size_t ws_size, hipStream_t stream) {
    (void)n_in; (void)out_size; (void)ws_size;
    const float* nfeats = (const float*)d_in[0];
    const float* efeats = (const float*)d_in[1];
    const float* W1a    = (const float*)d_in[2];
    const float* b1a    = (const float*)d_in[3];
    const float* W1e    = (const float*)d_in[4];
    const float* b1e    = (const float*)d_in[5];
    const float* W2a    = (const float*)d_in[6];
    const float* b2a    = (const float*)d_in[7];
    const float* W2e    = (const float*)d_in[8];
    const float* b2e    = (const float*)d_in[9];
    const int*   u      = (const int*)d_in[10];
    const int*   v      = (const int*)d_in[11];

    const int N = in_sizes[0] / 64;  // 50000
    const int E = in_sizes[10];      // 800000

    float* ws   = (float*)d_ws;
    float* deg  = ws;                        // N
    float* msum = deg + N;                   // N*256 max (layer1 uses N*128)
    float* hn1  = msum + (size_t)N * 256;    // N*128 fp32
    float* he1  = hn1 + (size_t)N * 128;     // E*128 fp32
    __bf16* hnb1 = (__bf16*)(he1 + (size_t)E * 128);  // N*128 bf16
    __bf16* hnb2 = hnb1 + (size_t)N * 128;            // N*128 bf16
    __bf16* W1at = hnb2 + (size_t)N * 128;            // 192*128 bf16 (transposed)
    __bf16* W1et = W1at + 192 * 128;                  // 256*128
    __bf16* W2at = W1et + 256 * 128;                  // 384*128
    __bf16* W2et = W2at + 384 * 128;                  // 256*128

    float* hn2 = (float*)d_out;              // N*128 (output part 1)
    float* he2 = hn2 + (size_t)N * 128;      // E*128 (output part 2)

    const int nodeBlocks = (N + NG_ROWS - 1) / NG_ROWS;
    const int edgeBlocks = (E + EG_EDGES - 1) / EG_EDGES;

    // ---- one-time prep: weights -> transposed bf16 ----
    convert_wt_kernel<<<96, 256, 0, stream>>>(W1a, W1at, 192);
    convert_wt_kernel<<<128, 256, 0, stream>>>(W1e, W1et, 256);
    convert_wt_kernel<<<192, 256, 0, stream>>>(W2a, W2at, 384);
    convert_wt_kernel<<<128, 256, 0, stream>>>(W2e, W2et, 256);

    // ---- layer 1 ----
    zero_kernel<<<2048, 256, 0, stream>>>(deg, (long)N + (long)N * 128);
    deg_kernel<<<2048, 256, 0, stream>>>(v, deg, E);
    scatter_kernel<<<8192, 256, 0, stream>>>(nfeats, 64, efeats, 64, u, v,
                                             msum, E);
    node_gemm_kernel<<<nodeBlocks, 256, 0, stream>>>(
        nfeats, 64, msum, 128, deg, W1at, b1a, hn1, hnb1, N);
    edge_gemm_kernel<<<edgeBlocks, 256, 0, stream>>>(hnb1, u, v, W1et, b1e,
                                                     he1, E);
    // ---- layer 2 ----
    zero_kernel<<<2048, 256, 0, stream>>>(msum, (long)N * 256);
    scatter_kernel<<<8192, 256, 0, stream>>>(hn1, 128, he1, 128, u, v, msum,
                                             E);
    node_gemm_kernel<<<nodeBlocks, 256, 0, stream>>>(
        hn1, 128, msum, 256, deg, W2at, b2a, hn2, hnb2, N);
    edge_gemm_kernel<<<edgeBlocks, 256, 0, stream>>>(hnb2, u, v, W2et, b2e,
                                                     he2, E);
}
